// HRqVae_90168543412482
// MI455X (gfx1250) — compile-verified, matrix operand order checked by reading, not run on hardware
//
#include <hip/hip_runtime.h>
#include <hip/hip_bf16.h>

// ---------------------------------------------------------------------------
// MI455X (gfx1250, wave32) HRQ-VAE forward.
// All matmuls: v_wmma_f32_16x16x32_bf16 (bf16 operands, f32 accumulate).
// B operands stored N-major (N x K) so every 16-element fragment is two
// contiguous 16-byte loads; A tiles staged in LDS with float4->v4bf packing.
// k_gemm: 32x16 tile per wave (2 accumulators share one B fragment).
// B x B alignment logits and B x 4096 codebook distances computed fused
// (online softmax / online argmin) -- nothing bigger than activations in HBM.
// ---------------------------------------------------------------------------

#define BB   8192
#define INF_ 768
#define DD   128
#define KC   4096
#define NLY  3
#define TAGD 768

typedef __attribute__((ext_vector_type(16))) __bf16 v16bf;
typedef __attribute__((ext_vector_type(8)))  __bf16 v8bf;
typedef __attribute__((ext_vector_type(4)))  __bf16 v4bf;
typedef __attribute__((ext_vector_type(8)))  float  v8f;

__device__ __forceinline__ __bf16 f2bf(float f) {
  union { float f; unsigned u; } v; v.f = f;
  unsigned r = v.u + 0x7FFFu + ((v.u >> 16) & 1u);   // round-to-nearest-even
  unsigned short h = (unsigned short)(r >> 16);
  return __builtin_bit_cast(__bf16, h);
}

// Build a 16-element bf16 fragment from a 32-long K-run at `base`:
// elements 0..7  <- k = 8*half + 0..7
// elements 8..15 <- k = 16 + 8*half + 0..7     (two contiguous 16B loads)
__device__ __forceinline__ v16bf frag_from(const __bf16* base, int half) {
  union { v16bf v; v8bf h[2]; } u;
  u.h[0] = *(const v8bf*)(base + 8 * half);
  u.h[1] = *(const v8bf*)(base + 16 + 8 * half);
  return u.v;
}

// ---------------------------------------------------------------------------
// GEMM: C[M=8192,N] = act(A @ W + bias).  A: f32 (stride lda, lda%4==0),
// Wt: bf16 N-major (rows padded to mult of 16), C: f32 (stride ldc).
// act: 0=none 1=relu 2=sigmoid.
// Block=256 (8 waves) covers 32 rows x 128 cols; each wave one 32x16 tile
// (two wmma sharing the B fragment). A staged 32x32 in LDS, one float4/thread.
// ---------------------------------------------------------------------------
__global__ void k_gemm(const float* __restrict__ A, int lda,
                       const __bf16* __restrict__ Wt,
                       const float* __restrict__ bias,
                       float* __restrict__ C, int ldc,
                       int N, int K, int act) {
  __shared__ __bf16 As[32 * 32];
  const int lane = threadIdx.x & 31;
  const int wave = threadIdx.x >> 5;
  const int half = lane >> 4;
  const int l15  = lane & 15;
  const int m0 = blockIdx.y * 32;
  const int n0 = blockIdx.x * 128 + wave * 16;
  const bool active_n = (n0 < N);                 // wave-uniform
  const __bf16* wrow = Wt + (long)(n0 + l15) * K; // padded rows: always safe

  const int sr  = threadIdx.x >> 3;               // 32 rows x 8 float4 chunks
  const int sc4 = (threadIdx.x & 7) << 2;
  const float* arow = A + (long)(m0 + sr) * lda + sc4;

  v8f c0 = {}, c1 = {};
  for (int k0 = 0; k0 < K; k0 += 32) {
    __syncthreads();
    {
      float4 f = *(const float4*)(arow + k0);
      v4bf p = { f2bf(f.x), f2bf(f.y), f2bf(f.z), f2bf(f.w) };
      *(v4bf*)&As[sr * 32 + sc4] = p;
    }
    __syncthreads();
    if (active_n) {
      if (k0 + 32 < K) __builtin_prefetch(wrow + k0 + 32, 0, 3);
      v16bf b  = frag_from(wrow + k0, half);
      v16bf a0 = frag_from(&As[l15 * 32], half);
      v16bf a1 = frag_from(&As[(16 + l15) * 32], half);
      c0 = __builtin_amdgcn_wmma_f32_16x16x32_bf16(false, a0, false, b,
                                                   (short)0, c0, false, false);
      c1 = __builtin_amdgcn_wmma_f32_16x16x32_bf16(false, a1, false, b,
                                                   (short)0, c1, false, false);
    }
  }
  if (!active_n) return;
  const int n = n0 + l15;
  if (n >= N) return;
  const float bi = bias[n];
#pragma unroll
  for (int r = 0; r < 8; ++r) {
    int m = m0 + r + (half << 3);
    float v = c0[r] + bi;
    if (act == 1)      v = v > 0.0f ? v : 0.0f;
    else if (act == 2) v = 1.0f / (1.0f + __expf(-v));
    C[(long)m * ldc + n] = v;
  }
#pragma unroll
  for (int r = 0; r < 8; ++r) {
    int m = m0 + 16 + r + (half << 3);
    float v = c1[r] + bi;
    if (act == 1)      v = v > 0.0f ? v : 0.0f;
    else if (act == 2) v = 1.0f / (1.0f + __expf(-v));
    C[(long)m * ldc + n] = v;
  }
}

// ---------------------------------------------------------------------------
// Fused codebook search: ids[b] = argmin_n (||cb_n||^2 - 2 * res_b . cb_n)
// res: B x 128 f32.  cbB: 4096 x 128 bf16 (N-major == original layout).
// ---------------------------------------------------------------------------
__global__ void k_quantize(const float* __restrict__ res,
                           const __bf16* __restrict__ cbB,
                           const float* __restrict__ cbn2,
                           int* __restrict__ ids) {
  __shared__ __bf16 As[16 * 128];
  __shared__ float bS[8][16];
  __shared__ int   bI[8][16];
  const int lane = threadIdx.x & 31, wave = threadIdx.x >> 5;
  const int half = lane >> 4, l15 = lane & 15;
  const int m0 = blockIdx.x * 16;

  {                                               // 16 x 128 f32 -> bf16 LDS
    int t = threadIdx.x;                          // 512 float4 chunks
    for (int i = 0; i < 2; ++i, t += 256) {
      int r  = t >> 5;
      int c4 = (t & 31) << 2;
      float4 f = *(const float4*)(res + (long)(m0 + r) * 128 + c4);
      v4bf p = { f2bf(f.x), f2bf(f.y), f2bf(f.z), f2bf(f.w) };
      *(v4bf*)&As[r * 128 + c4] = p;
    }
  }
  __syncthreads();

  v16bf afrag[4];
#pragma unroll
  for (int ks = 0; ks < 4; ++ks)
    afrag[ks] = frag_from(&As[l15 * 128 + ks * 32], half);

  float best[8]; int bidx[8];
#pragma unroll
  for (int r = 0; r < 8; ++r) { best[r] = 3.4e38f; bidx[r] = 0; }

  for (int nt = wave; nt < KC / 16; nt += 8) {
    int n0 = nt * 16;
    const __bf16* crow = cbB + (long)(n0 + l15) * 128;
    v8f c = {};
#pragma unroll
    for (int ks = 0; ks < 4; ++ks) {
      v16bf b = frag_from(crow + ks * 32, half);
      c = __builtin_amdgcn_wmma_f32_16x16x32_bf16(false, afrag[ks], false, b,
                                                  (short)0, c, false, false);
    }
    int n = n0 + l15;
    float cn = cbn2[n];
#pragma unroll
    for (int r = 0; r < 8; ++r) {
      float s = cn - 2.0f * c[r];
      if (s < best[r] || (s == best[r] && n < bidx[r])) { best[r] = s; bidx[r] = n; }
    }
  }
#pragma unroll
  for (int r = 0; r < 8; ++r) {
    float s = best[r]; int bi = bidx[r];
#pragma unroll
    for (int mask = 1; mask < 16; mask <<= 1) {
      float os = __shfl_xor(s, mask);
      int   oi = __shfl_xor(bi, mask);
      if (os < s || (os == s && oi < bi)) { s = os; bi = oi; }
    }
    if (l15 == 0) { int row = r + 8 * half; bS[wave][row] = s; bI[wave][row] = bi; }
  }
  __syncthreads();
  if (threadIdx.x < 16) {
    float s = bS[0][threadIdx.x]; int bi = bI[0][threadIdx.x];
    for (int w = 1; w < 8; ++w) {
      float os = bS[w][threadIdx.x]; int oi = bI[w][threadIdx.x];
      if (os < s || (os == s && oi < bi)) { s = os; bi = oi; }
    }
    ids[m0 + threadIdx.x] = bi;
  }
}

// ---------------------------------------------------------------------------
// Fused alignment loss: per row i of en (B x Di):
//   per_i = logsumexp_j(en_i . tn_j / T) - en_i . tn_i / T
// tnB: B x Di bf16 (N-major).  Online softmax over 16-wide column tiles.
// ---------------------------------------------------------------------------
__global__ void k_align(const float* __restrict__ en, int lda,
                        const __bf16* __restrict__ tnB,
                        int Di, float invT,
                        float* __restrict__ aloss, float scale) {
  extern __shared__ __bf16 EnS[];                 // 16 * Di
  __shared__ float rM[8][16], rS[8][16], rD[8][16];
  const int lane = threadIdx.x & 31, wave = threadIdx.x >> 5;
  const int half = lane >> 4, l15 = lane & 15;
  const int m0 = blockIdx.x * 16;
  const int Di4 = Di >> 2;

  for (int r = 0; r < 16; ++r) {
    for (int c4 = threadIdx.x; c4 < Di4; c4 += 256) {
      float4 f = *(const float4*)(en + (long)(m0 + r) * lda + (c4 << 2));
      v4bf p = { f2bf(f.x), f2bf(f.y), f2bf(f.z), f2bf(f.w) };
      *(v4bf*)&EnS[r * Di + (c4 << 2)] = p;
    }
  }
  __syncthreads();

  float mx[8], sm[8], dg[8];
#pragma unroll
  for (int r = 0; r < 8; ++r) { mx[r] = -3.4e38f; sm[r] = 0.0f; dg[r] = -3.4e38f; }

  const int nks = Di >> 5;
  for (int nt = wave; nt < BB / 16; nt += 8) {
    int n0 = nt * 16;
    const __bf16* trow = tnB + (long)(n0 + l15) * Di;
    v8f c = {};
    for (int ks = 0; ks < nks; ++ks) {
      v16bf a = frag_from(&EnS[l15 * Di + ks * 32], half);
      v16bf b = frag_from(trow + ks * 32, half);
      c = __builtin_amdgcn_wmma_f32_16x16x32_bf16(false, a, false, b,
                                                  (short)0, c, false, false);
    }
    int n = n0 + l15;
#pragma unroll
    for (int r = 0; r < 8; ++r) {
      float v = c[r] * invT;
      int m = m0 + r + (half << 3);
      if (n == m) dg[r] = v;
      float nm = fmaxf(mx[r], v);
      sm[r] = sm[r] * __expf(mx[r] - nm) + __expf(v - nm);
      mx[r] = nm;
    }
  }
#pragma unroll
  for (int r = 0; r < 8; ++r) {
    float m_ = mx[r], s_ = sm[r], d_ = dg[r];
#pragma unroll
    for (int mask = 1; mask < 16; mask <<= 1) {
      float om = __shfl_xor(m_, mask);
      float os = __shfl_xor(s_, mask);
      float od = __shfl_xor(d_, mask);
      float nm = fmaxf(m_, om);
      s_ = s_ * __expf(m_ - nm) + os * __expf(om - nm);
      m_ = nm;
      d_ = fmaxf(d_, od);
    }
    if (l15 == 0) {
      int row = r + 8 * half;
      rM[wave][row] = m_; rS[wave][row] = s_; rD[wave][row] = d_;
    }
  }
  __syncthreads();
  if (threadIdx.x < 16) {
    float m_ = rM[0][threadIdx.x], s_ = rS[0][threadIdx.x], d_ = rD[0][threadIdx.x];
    for (int w = 1; w < 8; ++w) {
      float om = rM[w][threadIdx.x], os = rS[w][threadIdx.x];
      float nm = fmaxf(m_, om);
      s_ = s_ * __expf(m_ - nm) + os * __expf(om - nm);
      m_ = nm;
      d_ = fmaxf(d_, rD[w][threadIdx.x]);
    }
    float per = (m_ + __logf(s_)) - d_;
    atomicAdd(aloss, per * scale);
  }
}

// ------------------------- VALU support kernels ----------------------------

__global__ void k_convert(const float* __restrict__ s, __bf16* __restrict__ d, long n) {
  long t = (long)blockIdx.x * blockDim.x + threadIdx.x;
  if (t < n) d[t] = f2bf(s[t]);
}

// src: R x C f32  ->  dst: C x R bf16  (dst rows beyond C unwritten/padded)
__global__ void k_convert_T(const float* __restrict__ s, __bf16* __restrict__ d,
                            int R, int C) {
  long t = (long)blockIdx.x * blockDim.x + threadIdx.x;
  if (t < (long)R * C) {
    int r = (int)(t / C), c = (int)(t - (long)r * C);
    d[(long)c * R + r] = f2bf(s[t]);
  }
}

__global__ void k_cbnorm(const float* __restrict__ cb, float* __restrict__ out,
                         int codes, int dim) {
  int c = blockIdx.x * blockDim.x + threadIdx.x;
  if (c < codes) {
    const float* p = cb + (long)c * dim;
    float s = 0.0f;
    for (int d = 0; d < dim; ++d) s += p[d] * p[d];
    out[c] = s;
  }
}

// per-row L2 normalize: out = x / max(||x||, eps)
__global__ void k_l2n(const float* __restrict__ in, int lda,
                      float* __restrict__ out, int ldo, int dim) {
  int row = blockIdx.x;
  __shared__ float red[256];
  float s = 0.0f;
  for (int d = threadIdx.x; d < dim; d += 256) {
    float v = in[(long)row * lda + d]; s += v * v;
  }
  red[threadIdx.x] = s; __syncthreads();
  for (int st = 128; st > 0; st >>= 1) {
    if (threadIdx.x < st) red[threadIdx.x] += red[threadIdx.x + st];
    __syncthreads();
  }
  float inv = 1.0f / fmaxf(sqrtf(red[0]), 1e-12f);
  for (int d = threadIdx.x; d < dim; d += 256)
    out[(long)row * ldo + d] = in[(long)row * lda + d] * inv;
}

// layernorm with affine, act: 0=none 1=relu (in-place safe)
__global__ void k_layernorm(const float* __restrict__ in, float* __restrict__ out,
                            const float* __restrict__ g, const float* __restrict__ b,
                            int dim, int act) {
  int row = blockIdx.x;
  __shared__ float r1[256], r2[256];
  float s = 0.0f, s2 = 0.0f;
  for (int d = threadIdx.x; d < dim; d += 256) {
    float v = in[(long)row * dim + d]; s += v; s2 += v * v;
  }
  r1[threadIdx.x] = s; r2[threadIdx.x] = s2; __syncthreads();
  for (int st = 128; st > 0; st >>= 1) {
    if (threadIdx.x < st) { r1[threadIdx.x] += r1[threadIdx.x + st]; r2[threadIdx.x] += r2[threadIdx.x + st]; }
    __syncthreads();
  }
  float mean = r1[0] / dim;
  float var  = r2[0] / dim - mean * mean;
  float inv  = rsqrtf(var + 1e-5f);
  for (int d = threadIdx.x; d < dim; d += 256) {
    float v = (in[(long)row * dim + d] - mean) * inv * g[d] + b[d];
    if (act == 1) v = v > 0.0f ? v : 0.0f;
    out[(long)row * dim + d] = v;
  }
}

__global__ void k_affine_relu(float* __restrict__ x, const float* __restrict__ g,
                              const float* __restrict__ b, long n, int cols) {
  long t = (long)blockIdx.x * blockDim.x + threadIdx.x;
  if (t < n) {
    int c = (int)(t % cols);
    float v = x[t] * g[c] + b[c];
    x[t] = v > 0.0f ? v : 0.0f;
  }
}

__global__ void k_mul(const float* __restrict__ a, int lda,
                      const float* __restrict__ b, int ldb,
                      float* __restrict__ o, int ldo, int cols) {
  long t = (long)blockIdx.x * blockDim.x + threadIdx.x;
  if (t < (long)BB * cols) {
    int row = (int)(t / cols), c = (int)(t - (long)row * cols);
    o[(long)row * ldo + c] = a[(long)row * lda + c] * b[(long)row * ldb + c];
  }
}

__global__ void k_add_inplace(float* __restrict__ acc, const float* __restrict__ x, long n) {
  long t = (long)blockIdx.x * blockDim.x + threadIdx.x;
  if (t < n) acc[t] += x[t];
}

// quantizer update: write outputs, straight-through residual, qloss partial
__global__ void k_quant_update(float* __restrict__ res, const float* __restrict__ cb,
                               const int* __restrict__ ids,
                               float* __restrict__ out_embs, float* __restrict__ out_res,
                               float* __restrict__ out_ids, float* __restrict__ concat,
                               int layer, float* __restrict__ qloss, float qscale) {
  long t = (long)blockIdx.x * blockDim.x + threadIdx.x;
  __shared__ float red[256];
  float sq = 0.0f;
  if (t < (long)BB * DD) {
    int b = (int)(t >> 7), d = (int)(t & 127);
    int id = ids[b];
    float e = cb[(long)id * DD + d];
    float r = res[t];
    out_res [t * NLY + layer] = r;          // residuals[b,d,l]
    out_embs[t * NLY + layer] = e;          // embs[b,d,l] (straight-through value)
    concat[(long)b * (DD * NLY) + layer * DD + d] = e;
    res[t] = r - e;
    float df = r - e;
    sq = df * df;
    if (d == 0) out_ids[(long)b * NLY + layer] = (float)id;
  }
  red[threadIdx.x] = sq; __syncthreads();
  for (int st = 128; st > 0; st >>= 1) {
    if (threadIdx.x < st) red[threadIdx.x] += red[threadIdx.x + st];
    __syncthreads();
  }
  if (threadIdx.x == 0) atomicAdd(qloss, red[0] * qscale);
}

// sum of the 3 layer embeddings from the interleaved [b,d,l] output buffer
__global__ void k_sum_embs(const float* __restrict__ embs, float* __restrict__ out) {
  long t = (long)blockIdx.x * blockDim.x + threadIdx.x;
  if (t < (long)BB * DD)
    out[t] = embs[t * NLY + 0] + embs[t * NLY + 1] + embs[t * NLY + 2];
}

// row softmax cross-entropy at label + argmax accuracy
__global__ void k_ce(const float* __restrict__ logits, int nc,
                     const int* __restrict__ labels, int layer,
                     float* __restrict__ ploss, float* __restrict__ pacc, float scale) {
  int row = blockIdx.x;
  __shared__ float rv[256]; __shared__ int ri[256];
  const float* lr = logits + (long)row * nc;
  float mx = -3.4e38f; int mi = nc;
  for (int c = threadIdx.x; c < nc; c += 256) {
    float v = lr[c];
    if (v > mx || (v == mx && c < mi)) { mx = v; mi = c; }
  }
  rv[threadIdx.x] = mx; ri[threadIdx.x] = mi; __syncthreads();
  for (int st = 128; st > 0; st >>= 1) {
    if (threadIdx.x < st) {
      float ov = rv[threadIdx.x + st]; int oi = ri[threadIdx.x + st];
      if (ov > rv[threadIdx.x] || (ov == rv[threadIdx.x] && oi < ri[threadIdx.x])) {
        rv[threadIdx.x] = ov; ri[threadIdx.x] = oi;
      }
    }
    __syncthreads();
  }
  mx = rv[0]; mi = ri[0];
  __syncthreads();
  float s = 0.0f;
  for (int c = threadIdx.x; c < nc; c += 256) s += __expf(lr[c] - mx);
  rv[threadIdx.x] = s; __syncthreads();
  for (int st = 128; st > 0; st >>= 1) {
    if (threadIdx.x < st) rv[threadIdx.x] += rv[threadIdx.x + st];
    __syncthreads();
  }
  if (threadIdx.x == 0) {
    float lse = mx + __logf(rv[0]);
    int lab = labels[(long)row * NLY + layer];
    atomicAdd(ploss, (lse - lr[lab]) * scale);
    atomicAdd(pacc, (mi == lab) ? scale : 0.0f);
  }
}

__global__ void k_zero(float* __restrict__ p, int n) {
  int t = blockIdx.x * blockDim.x + threadIdx.x;
  if (t < n) p[t] = 0.0f;
}

// ---------------------------------------------------------------------------
// Host orchestration
// ---------------------------------------------------------------------------
struct Lin { const float* b; const float* w; };
struct Ln  { const float* b; const float* g; };

extern "C" void kernel_launch(void* const* d_in, const int* in_sizes, int n_in,
                              void* d_out, int out_size, void* d_ws, size_t ws_size,
                              hipStream_t stream) {
  (void)in_sizes; (void)out_size; (void)ws_size; (void)n_in;

  // ---- inputs (setup_inputs order, params pytree in sorted-key order) ----
  const float* x_in      = (const float*)d_in[0];   // B x 768
  const float* tags_emb  = (const float*)d_in[1];   // B x NL x 768
  const int*   tags_idx  = (const int*)  d_in[2];   // B x NL
  const float* codebooks = (const float*)d_in[3];   // NL x 4096 x 128
  int p = 4;
  auto nxt = [&]() { return (const float*)d_in[p++]; };

  Lin dec[4], enc[4];
  for (int i = 0; i < 4; ++i) { dec[i].b = nxt(); dec[i].w = nxt(); }
  for (int i = 0; i < 4; ++i) { enc[i].b = nxt(); enc[i].w = nxt(); }
  // pred[i] keys sorted: a1,a2,ca,ca_ln,cb,fe,fe_ln,r1a,r1a_ln,r1b,r1b_ln,r2a,r2a_ln,r2b,r2b_ln
  Lin a1[3], a2[3], ca[3], cbl[3], fe[3], r1a[3], r1b[3], r2a[3], r2b[3];
  Ln  ca_ln[3], fe_ln[3], r1a_ln[3], r1b_ln[3], r2a_ln[3], r2b_ln[3];
  for (int i = 0; i < 3; ++i) {
    a1[i].b = nxt(); a1[i].w = nxt();
    a2[i].b = nxt(); a2[i].w = nxt();
    ca[i].b = nxt(); ca[i].w = nxt();
    ca_ln[i].b = nxt(); ca_ln[i].g = nxt();
    cbl[i].b = nxt(); cbl[i].w = nxt();
    fe[i].b = nxt(); fe[i].w = nxt();
    fe_ln[i].b = nxt(); fe_ln[i].g = nxt();
    r1a[i].b = nxt(); r1a[i].w = nxt();
    r1a_ln[i].b = nxt(); r1a_ln[i].g = nxt();
    r1b[i].b = nxt(); r1b[i].w = nxt();
    r1b_ln[i].b = nxt(); r1b_ln[i].g = nxt();
    r2a[i].b = nxt(); r2a[i].w = nxt();
    r2a_ln[i].b = nxt(); r2a_ln[i].g = nxt();
    r2b[i].b = nxt(); r2b[i].w = nxt();
    r2b_ln[i].b = nxt(); r2b_ln[i].g = nxt();
  }
  Ln bn[3]; Lin pl1[3], pl2[3];
  for (int i = 0; i < 3; ++i) {
    bn[i].b = nxt(); bn[i].g = nxt();
    pl1[i].b = nxt(); pl1[i].w = nxt();
    pl2[i].b = nxt(); pl2[i].w = nxt();
  }

  // ---- workspace bump allocator ----
  char* wsp = (char*)d_ws;
  size_t off = 0;
  auto alloc = [&](size_t bytes) -> void* {
    void* r = wsp + off; off += (bytes + 255) & ~(size_t)255; return r;
  };
  const int NELT = 256;
  auto ew = [&](long n) { return dim3((unsigned)((n + NELT - 1) / NELT)); };
  // weights: K x N f32 -> N-major bf16, rows padded to mult of 16
  auto convT = [&](const float* w, int K, int N) -> __bf16* {
    int Np = (N + 15) & ~15;
    __bf16* dst = (__bf16*)alloc((size_t)Np * K * sizeof(__bf16));
    k_convert_T<<<ew((long)K * N), NELT, 0, stream>>>(w, dst, K, N);
    return dst;
  };

  float* res    = (float*)alloc((size_t)BB * DD * 4);
  float* concat = (float*)alloc((size_t)BB * (DD * NLY) * 4);
  float* bufA   = (float*)alloc((size_t)BB * 768 * 4);
  float* bufB   = (float*)alloc((size_t)BB * 768 * 4);
  float* bufC   = (float*)alloc((size_t)BB * 768 * 4);
  float* bufD   = (float*)alloc((size_t)BB * 768 * 4);
  float* bufE   = (float*)alloc((size_t)BB * 1024 * 4);  // tag logits (nc<=1000)
  __bf16* cbB   = (__bf16*)alloc((size_t)KC * DD * 2);   // codebook bf16 (N-major)
  __bf16* tnB   = (__bf16*)alloc((size_t)BB * 384 * 2);  // tn bf16 (N-major)
  float*  cbn2  = (float*)alloc((size_t)KC * 4);
  int*    ids   = (int*)alloc((size_t)BB * 4);

  // ---- output layout (flat, return order) ----
  float* out      = (float*)d_out;
  const long O_EMB  = 0;
  const long O_RES  = (long)BB * DD * NLY;
  const long O_IDS  = 2L * BB * DD * NLY;
  const long O_XHAT = O_IDS + (long)BB * NLY;
  const long O_Q    = O_XHAT + (long)BB * INF_;
  float* qptr = out + O_Q;       // qloss, aloss, ploss, pacc
  k_zero<<<1, 32, 0, stream>>>(qptr, 4);

  auto gemm = [&](const float* A, int lda, const __bf16* Wt, const float* bias,
                  float* C, int ldc, int N, int K, int act) {
    dim3 g((unsigned)((N + 127) / 128), BB / 32);
    k_gemm<<<g, 256, 0, stream>>>(A, lda, Wt, bias, C, ldc, N, K, act);
  };

  // ---- weights -> bf16 N-major ----
  static const int encK[4] = {768, 512, 256, 128}, encN[4] = {512, 256, 128, 128};
  static const int decK[4] = {128, 128, 256, 512}, decN[4] = {128, 256, 512, 768};
  __bf16 *enc_wb[4], *dec_wb[4];
  for (int i = 0; i < 4; ++i) enc_wb[i] = convT(enc[i].w, encK[i], encN[i]);
  for (int i = 0; i < 4; ++i) dec_wb[i] = convT(dec[i].w, decK[i], decN[i]);

  // ---- encoder: x -> res ----
  gemm(x_in, 768, enc_wb[0], enc[0].b, bufA, 512, 512, 768, 1);
  gemm(bufA, 512, enc_wb[1], enc[1].b, bufB, 256, 256, 512, 1);
  gemm(bufB, 256, enc_wb[2], enc[2].b, bufC, 128, 128, 256, 1);
  gemm(bufC, 128, enc_wb[3], enc[3].b, res,  128, 128, 128, 0);

  const float qscale = (1.0f + 0.25f) / ((float)BB * DD);
  const float ascale = 0.5f / ((float)BB * NLY);
  const float cscale = 1.0f / ((float)BB * NLY);

  for (int l = 0; l < NLY; ++l) {
    const int Di = DD * (l + 1);
    const int Hi = 256 * (l + 1);
    const int Mi = 256;
    const int nc = (l == 0) ? 10 : (l == 1) ? 100 : 1000;
    const float* cb_l = codebooks + (long)l * KC * DD;

    // -- quantize --
    k_convert<<<ew((long)KC * DD), NELT, 0, stream>>>(cb_l, cbB, (long)KC * DD);
    k_cbnorm<<<dim3(KC / 256), 256, 0, stream>>>(cb_l, cbn2, KC, DD);
    k_quantize<<<dim3(BB / 16), 256, 0, stream>>>(res, cbB, cbn2, ids);
    k_quant_update<<<ew((long)BB * DD), NELT, 0, stream>>>(
        res, cb_l, ids, out + O_EMB, out + O_RES, out + O_IDS, concat, l,
        qptr + 0, qscale);

    // -- projector: tags_emb[:,l] -> tn --
    __bf16* wl1 = convT(pl1[l].w, TAGD, 512);
    __bf16* wl2 = convT(pl2[l].w, 512, Di);
    gemm(tags_emb + (long)l * TAGD, NLY * TAGD, wl1, pl1[l].b, bufA, 512, 512, TAGD, 0);
    k_affine_relu<<<ew((long)BB * 512), NELT, 0, stream>>>(bufA, bn[l].g, bn[l].b,
                                                          (long)BB * 512, 512);
    gemm(bufA, 512, wl2, pl2[l].b, bufB, Di, Di, 512, 0);
    k_l2n<<<dim3(BB), 256, 0, stream>>>(bufB, Di, bufB, Di, Di);      // tn
    k_convert<<<ew((long)BB * Di), NELT, 0, stream>>>(bufB, tnB, (long)BB * Di);

    // -- alignment loss (fused online softmax over B x B) --
    k_l2n<<<dim3(BB), 256, 0, stream>>>(concat, DD * NLY, bufD, Di, Di);  // en
    k_align<<<dim3(BB / 16), 256, (size_t)16 * Di * 2, stream>>>(
        bufD, Di, tnB, Di, 1.0f / 0.1f, qptr + 1, ascale);

    // -- tag predictor --
    __bf16* wa1 = convT(a1[l].w, Di, Di / 4);
    __bf16* wa2 = convT(a2[l].w, Di / 4, Di);
    __bf16* wfe = convT(fe[l].w, Di, Hi);
    __bf16* w1a = convT(r1a[l].w, Hi, Mi);
    __bf16* w1b = convT(r1b[l].w, Mi, Hi);
    __bf16* w2a = convT(r2a[l].w, Hi, Mi);
    __bf16* w2b = convT(r2b[l].w, Mi, Hi);
    __bf16* wca = convT(ca[l].w, Hi, Mi);
    __bf16* wcb = convT(cbl[l].w, Mi, nc);

    gemm(concat, DD * NLY, wa1, a1[l].b, bufA, Di / 4, Di / 4, Di, 1);
    gemm(bufA, Di / 4, wa2, a2[l].b, bufB, Di, Di, Di / 4, 2);        // att
    k_mul<<<ew((long)BB * Di), NELT, 0, stream>>>(concat, DD * NLY, bufB, Di,
                                                  bufB, Di, Di);      // xa
    if (l > 0) k_l2n<<<dim3(BB), 256, 0, stream>>>(bufB, Di, bufB, Di, Di);
    gemm(bufB, Di, wfe, fe[l].b, bufC, Hi, Hi, Di, 0);
    k_layernorm<<<dim3(BB), 256, 0, stream>>>(bufC, bufC, fe_ln[l].g, fe_ln[l].b, Hi, 1);
    // residual block 1
    gemm(bufC, Hi, w1a, r1a[l].b, bufA, Mi, Mi, Hi, 0);
    k_layernorm<<<dim3(BB), 256, 0, stream>>>(bufA, bufA, r1a_ln[l].g, r1a_ln[l].b, Mi, 1);
    gemm(bufA, Mi, w1b, r1b[l].b, bufB, Hi, Hi, Mi, 0);
    k_layernorm<<<dim3(BB), 256, 0, stream>>>(bufB, bufB, r1b_ln[l].g, r1b_ln[l].b, Hi, 0);
    k_add_inplace<<<ew((long)BB * Hi), NELT, 0, stream>>>(bufC, bufB, (long)BB * Hi);
    // residual block 2
    gemm(bufC, Hi, w2a, r2a[l].b, bufA, Mi, Mi, Hi, 0);
    k_layernorm<<<dim3(BB), 256, 0, stream>>>(bufA, bufA, r2a_ln[l].g, r2a_ln[l].b, Mi, 1);
    gemm(bufA, Mi, w2b, r2b[l].b, bufB, Hi, Hi, Mi, 0);
    k_layernorm<<<dim3(BB), 256, 0, stream>>>(bufB, bufB, r2b_ln[l].g, r2b_ln[l].b, Hi, 0);
    k_add_inplace<<<ew((long)BB * Hi), NELT, 0, stream>>>(bufC, bufB, (long)BB * Hi);
    // classifier
    gemm(bufC, Hi, wca, ca[l].b, bufA, Mi, Mi, Hi, 0);
    k_layernorm<<<dim3(BB), 256, 0, stream>>>(bufA, bufA, ca_ln[l].g, ca_ln[l].b, Mi, 1);
    gemm(bufA, Mi, wcb, cbl[l].b, bufE, nc, nc, Mi, 0);
    k_ce<<<dim3(BB), 256, 0, stream>>>(bufE, nc, tags_idx, l, qptr + 2, qptr + 3, cscale);
  }

  // ---- decoder: x_hat = mlp(l2n(sum(embs))) ----
  k_sum_embs<<<ew((long)BB * DD), NELT, 0, stream>>>(out + O_EMB, bufA);
  k_l2n<<<dim3(BB), 256, 0, stream>>>(bufA, 128, bufA, 128, 128);
  gemm(bufA, 128, dec_wb[0], dec[0].b, bufB, 128, 128, 128, 1);
  gemm(bufB, 128, dec_wb[1], dec[1].b, bufC, 256, 256, 128, 1);
  gemm(bufC, 256, dec_wb[2], dec[2].b, bufA, 512, 512, 256, 1);
  gemm(bufA, 512, dec_wb[3], dec[3].b, out + O_XHAT, 768, 768, 512, 0);
}
